// AttentionModule_66537633349985
// MI455X (gfx1250) — matrix operand order repaired and checked by reading, compiled, not run on hardware
//
#include <hip/hip_runtime.h>
#include <hip/hip_bf16.h>
#include <math.h>

typedef __attribute__((ext_vector_type(16))) __bf16 v16bf;
typedef __attribute__((ext_vector_type(8)))  float  v8f;

#define HDIM 128
#define WSTRIDE 136   // pad 8 bf16 per row -> 4-dword bank shift per row

// ---------------------------------------------------------------------------
// Kernel 1: scores[n] = tanh(E[n,:] @ Wc^T + bc) . wa + ba
// 128 threads = 4 waves; each wave owns a 16-node tile; block covers 64 nodes.
// Wc staged in LDS as bf16 once per block. 32 v_wmma per wave.
// ---------------------------------------------------------------------------
__global__ __launch_bounds__(128) void score_kernel(
    const float* __restrict__ emb, const float* __restrict__ Wc,
    const float* __restrict__ bc,  const float* __restrict__ wa,
    const float* __restrict__ ba,  float* __restrict__ scores, int N)
{
    __shared__ __bf16 Wlds[HDIM * WSTRIDE];
    const int tid = threadIdx.x;

    // Stage Wc (128x128 f32 -> bf16 LDS), fully coalesced
    for (int i = tid; i < HDIM * HDIM; i += 128) {
        int r = i >> 7, c = i & 127;
        Wlds[r * WSTRIDE + c] = (__bf16)Wc[i];
    }
    __syncthreads();

    const int wave = tid >> 5;
    const int lane = tid & 31;
    const int half = lane >> 4;   // 0: lanes 0-15, 1: lanes 16-31
    const int l15  = lane & 15;

    const int rowBase = (blockIdx.x * 4 + wave) * 16;
    if (rowBase >= N) return;     // wave-uniform exit (EXEC stays all-1)

    v8f acc[8];
#pragma unroll
    for (int j = 0; j < 8; ++j)
        acc[j] = (v8f){0.f,0.f,0.f,0.f,0.f,0.f,0.f,0.f};

    const int arow = min(rowBase + l15, N - 1);       // keep all lanes active
    const float* erow = emb + (size_t)arow * HDIM;

#pragma unroll
    for (int kc = 0; kc < 4; ++kc) {
        // A fragment (16x32 bf16): lane row M=l15 holds K = k0..k0+15
        const int k0 = kc * 32 + half * 16;
        v16bf a;
#pragma unroll
        for (int e = 0; e < 8; ++e) a[e]     = (__bf16)erow[k0 + e];
#pragma unroll
        for (int e = 0; e < 8; ++e) a[8 + e] = (__bf16)erow[k0 + 8 + e];

        const int kb = kc * 32 + half * 16;
#pragma unroll
        for (int j = 0; j < 8; ++j) {
            // B fragment (32x16): lane col N=l15 holds K = kb..kb+15;
            // B[k][n] = Wc[j*16+n][k]  (transformed = E @ Wc^T)
            const __bf16* wrow = &Wlds[(j * 16 + l15) * WSTRIDE + kb];
            v16bf b;
#pragma unroll
            for (int e = 0; e < 16; ++e) b[e] = wrow[e];

            acc[j] = __builtin_amdgcn_wmma_f32_16x16x32_bf16(
                false, a, false, b, (short)0, acc[j], false, false);
        }
    }

    // Epilogue: lane l holds column c=j*16+l15, rows m = v + half*8 (v=0..7)
    float sc[8];
#pragma unroll
    for (int v = 0; v < 8; ++v) sc[v] = 0.f;
#pragma unroll
    for (int j = 0; j < 8; ++j) {
        const int   c   = j * 16 + l15;
        const float bcc = bc[c];
        const float wac = wa[c];
#pragma unroll
        for (int v = 0; v < 8; ++v) {
            float t = tanhf(acc[j][v] + bcc);
            sc[v] += t * wac;
        }
    }
    // Sum over the 16 columns held by each half-wave (xor masks < 16 stay in half)
#pragma unroll
    for (int v = 0; v < 8; ++v) {
        sc[v] += __shfl_xor(sc[v], 8, 32);
        sc[v] += __shfl_xor(sc[v], 4, 32);
        sc[v] += __shfl_xor(sc[v], 2, 32);
        sc[v] += __shfl_xor(sc[v], 1, 32);
    }
    if (l15 == 0) {
        const float b_a = ba[0];
#pragma unroll
        for (int v = 0; v < 8; ++v) {
            int row = rowBase + v + half * 8;
            if (row < N) scores[row] = sc[v] + b_a;
        }
    }
}

// ---------------------------------------------------------------------------
// Kernel 2: one block per (sorted) segment. Binary-search bounds, block
// max / sum-exp, then stream E once: out[seg,h] = sum_n attn[n] * E[n,h].
// Thread t owns feature h=t -> row reads are fully coalesced (512B/node).
// ---------------------------------------------------------------------------
__global__ __launch_bounds__(128) void pool_kernel(
    const float* __restrict__ emb, const int* __restrict__ batch,
    const float* __restrict__ scores, float* __restrict__ out, int N)
{
    __shared__ float red[128];
    __shared__ float wbuf[128];
    __shared__ int   bounds[2];

    const int t   = threadIdx.x;
    const int seg = blockIdx.x;

    if (t < 2) {
        int key = seg + t, lo = 0, hi = N;
        while (lo < hi) { int mid = (lo + hi) >> 1;
                          if (batch[mid] < key) lo = mid + 1; else hi = mid; }
        bounds[t] = lo;
    }
    __syncthreads();
    const int start = bounds[0], end = bounds[1];

    if (start >= end) { out[seg * HDIM + t] = 0.f; return; }  // uniform branch

    // segment max
    float m = -INFINITY;
    for (int i = start + t; i < end; i += 128) m = fmaxf(m, scores[i]);
    red[t] = m; __syncthreads();
    for (int s = 64; s >= 1; s >>= 1) {
        if (t < s) red[t] = fmaxf(red[t], red[t + s]);
        __syncthreads();
    }
    m = red[0]; __syncthreads();

    // segment sum of exp
    float sum = 0.f;
    for (int i = start + t; i < end; i += 128) sum += expf(scores[i] - m);
    red[t] = sum; __syncthreads();
    for (int s = 64; s >= 1; s >>= 1) {
        if (t < s) red[t] += red[t + s];
        __syncthreads();
    }
    const float inv = 1.0f / red[0];
    __syncthreads();

    // weighted pooling; weights for a 128-node chunk staged in LDS
    float acc = 0.f;
    for (int c0 = start; c0 < end; c0 += 128) {
        int idx = c0 + t;
        wbuf[t] = (idx < end) ? expf(scores[idx] - m) * inv : 0.f;
        __syncthreads();
        int nn = min(128, end - c0);
        if (c0 + 128 < end)
            __builtin_prefetch(&emb[(size_t)(c0 + 128) * HDIM + t], 0, 0);
        for (int i = 0; i < nn; ++i)
            acc += wbuf[i] * emb[(size_t)(c0 + i) * HDIM + t];
        __syncthreads();
    }
    out[seg * HDIM + t] = acc;
}

// ---------------------------------------------------------------------------
extern "C" void kernel_launch(void* const* d_in, const int* in_sizes, int n_in,
                              void* d_out, int out_size, void* d_ws, size_t ws_size,
                              hipStream_t stream)
{
    const float* emb   = (const float*)d_in[0];
    const int*   batch = (const int*)  d_in[1];
    const float* Wc    = (const float*)d_in[2];
    const float* bc    = (const float*)d_in[3];
    const float* wa    = (const float*)d_in[4];
    const float* ba    = (const float*)d_in[5];
    float*       out   = (float*)d_out;

    const int N = in_sizes[0] / HDIM;   // 1,000,000
    const int B = out_size   / HDIM;    // 1024

    float* scores = (float*)d_ws;       // N floats of scratch

    dim3 gridA((N + 63) / 64);          // 4 waves/block * 16 nodes/wave
    score_kernel<<<gridA, 128, 0, stream>>>(emb, Wc, bc, wa, ba, scores, N);
    pool_kernel<<<dim3(B), 128, 0, stream>>>(emb, batch, scores, out, N);
}
